// MambaCopy_84439057039721
// MI455X (gfx1250) — matrix operand order repaired and checked by reading, not compiled, not run on hardware
//
#include <hip/hip_runtime.h>
#include <hip/hip_bf16.h>

typedef __attribute__((ext_vector_type(16))) _Float16 v16h;
typedef __attribute__((ext_vector_type(8)))  _Float16 h8;
typedef __attribute__((ext_vector_type(8)))  float    v8f;

#define BB 4
#define LL 1024
#define DMODEL 256
#define DINNER 512
#define NSTATE 128
#define DTRANK 16
#define VOCAB 20
#define MROWS (BB*LL)            // 4096
#define DBLW 272                 // dt_rank + 2*N
#define DBLP 288                 // DBL padded to multiple of 32 columns

__device__ __forceinline__ v16h load_afrag(const _Float16* ap) {
    // A 16x32 f16: lanes0-15 K=[0..7]+[16..23]; lanes16-31 shifted by 8 (hi*8 in ap)
    h8 lo = *(const h8*)ap;
    h8 hi = *(const h8*)(ap + 16);
    return __builtin_shufflevector(lo, hi, 0,1,2,3,4,5,6,7,8,9,10,11,12,13,14,15);
}
__device__ __forceinline__ v16h load_bfrag(const _Float16* bp) {
    // B 32x16 f16: lanes0-15 K=[0..7]+[16..23]; lanes16-31 shifted by 8 (hi*8 in bp)
    h8 lo = *(const h8*)bp;
    h8 hi = *(const h8*)(bp + 16);
    return __builtin_shufflevector(lo, hi, 0,1,2,3,4,5,6,7,8,9,10,11,12,13,14,15);
}

// ---------------------------------------------------------------------------
// WMMA GEMM: Out[M,N] = A[M,K] * W[N,K]^T (+ Res[M,N]); f16 inputs, f32 accum.
// One wave computes a 64(M) x 32(N) macro-tile: 8 accumulators, A frags reused
// 2x, B frags reused 4x (12 b128 loads / 8 WMMAs). Fragments double-buffered so
// next-K loads are in flight during the current WMMA batch. K compile-time.
// 8 waves/block. Requires M%64==0, N%32==0, K%32==0.
// ---------------------------------------------------------------------------
template<int K>
__global__ __launch_bounds__(256)
void wmma_gemm_kernel(const _Float16* __restrict__ A, const _Float16* __restrict__ W,
                      const float* __restrict__ Res, float* __restrict__ Out,
                      int M, int N) {
    const int lane = threadIdx.x & 31;
    const int wave = threadIdx.x >> 5;
    const int ntn  = N >> 5;
    const int tile = blockIdx.x * 8 + wave;
    if (tile >= (M >> 6) * ntn) return;          // wave-uniform: EXEC stays all-ones
    const int tm  = tile / ntn;                  // 64-row macro tile
    const int tn  = tile - tm * ntn;             // 32-col macro tile
    const int col = lane & 15;                   // M-row (A) / N-col (B,C,D)
    const int hi  = lane >> 4;

    v8f c[4][2];
    if (Res) {
#pragma unroll
        for (int s = 0; s < 4; ++s)
#pragma unroll
            for (int t = 0; t < 2; ++t) {
                const float* rp = Res + (size_t)(tm*64 + s*16 + hi*8) * N + tn*32 + t*16 + col;
#pragma unroll
                for (int r = 0; r < 8; ++r) c[s][t][r] = rp[(size_t)r * N];
            }
    } else {
#pragma unroll
        for (int s = 0; s < 4; ++s)
#pragma unroll
            for (int t = 0; t < 2; ++t)
#pragma unroll
                for (int r = 0; r < 8; ++r) c[s][t][r] = 0.0f;
    }

    const _Float16* arow  = A + (size_t)(tm * 64 + col) * K + hi * 8;
    const _Float16* brow0 = W + (size_t)(tn * 32 + col) * K + hi * 8;
    const _Float16* brow1 = brow0 + (size_t)16 * K;

    v16h a[2][4], b[2][2];
    // prologue: load k0 = 0 fragments into buffer 0
#pragma unroll
    for (int s = 0; s < 4; ++s) a[0][s] = load_afrag(arow + (size_t)s * 16 * K);
    b[0][0] = load_bfrag(brow0);
    b[0][1] = load_bfrag(brow1);

#pragma unroll
    for (int k0 = 0; k0 < K; k0 += 32) {
        const int buf = (k0 >> 5) & 1;
        const int nxt = buf ^ 1;
        if (k0 + 32 < K) {   // issue next-K loads before consuming current buffer
#pragma unroll
            for (int s = 0; s < 4; ++s)
                a[nxt][s] = load_afrag(arow + (size_t)s * 16 * K + k0 + 32);
            b[nxt][0] = load_bfrag(brow0 + k0 + 32);
            b[nxt][1] = load_bfrag(brow1 + k0 + 32);
        }
#pragma unroll
        for (int s = 0; s < 4; ++s)
#pragma unroll
            for (int t = 0; t < 2; ++t)
                c[s][t] = __builtin_amdgcn_wmma_f32_16x16x32_f16(
                    false, a[buf][s], false, b[buf][t], (short)0, c[s][t], false, false);
    }

#pragma unroll
    for (int s = 0; s < 4; ++s)
#pragma unroll
        for (int t = 0; t < 2; ++t) {
            float* op = Out + (size_t)(tm*64 + s*16 + hi*8) * N + tn*32 + t*16 + col;
#pragma unroll
            for (int r = 0; r < 8; ++r) op[(size_t)r * N] = c[s][t][r];
        }
}

// ---------------------------------------------------------------------------
// f32 -> f16 convert with zero tail padding (for N-padded weights)
// ---------------------------------------------------------------------------
__global__ __launch_bounds__(256)
void cvt_f16_kernel(const float* __restrict__ in, _Float16* __restrict__ out,
                    int n_valid, int n_total) {
    const int i = blockIdx.x * 256 + threadIdx.x;
    if (i < n_total) out[i] = (i < n_valid) ? (_Float16)in[i] : (_Float16)0.0f;
}

// ---------------------------------------------------------------------------
// Embedding gather + LayerNorm; dual f32 + f16 output
// ---------------------------------------------------------------------------
__global__ __launch_bounds__(256)
void embed_ln_kernel(const int* __restrict__ tok, const float* __restrict__ emb,
                     const float* __restrict__ g, const float* __restrict__ bta,
                     float* __restrict__ Out, _Float16* __restrict__ OutH) {
    __shared__ float red[256];
    const int m = blockIdx.x, t = threadIdx.x;
    const float v = emb[tok[m] * DMODEL + t];
    red[t] = v; __syncthreads();
    for (int s = 128; s > 0; s >>= 1) { if (t < s) red[t] += red[t + s]; __syncthreads(); }
    const float mean = red[0] * (1.0f / DMODEL); __syncthreads();
    const float dv = v - mean;
    red[t] = dv * dv; __syncthreads();
    for (int s = 128; s > 0; s >>= 1) { if (t < s) red[t] += red[t + s]; __syncthreads(); }
    const float rstd = rsqrtf(red[0] * (1.0f / DMODEL) + 1e-5f);
    const float o = dv * rstd * g[t] + bta[t];
    Out [(size_t)m * DMODEL + t] = o;
    OutH[(size_t)m * DMODEL + t] = (_Float16)o;
}

__global__ __launch_bounds__(256)
void ln_kernel(const float* __restrict__ X, const float* __restrict__ g,
               const float* __restrict__ bta, float* __restrict__ Out,
               _Float16* __restrict__ OutH) {
    __shared__ float red[256];
    const int m = blockIdx.x, t = threadIdx.x;
    const float v = X[(size_t)m * DMODEL + t];
    red[t] = v; __syncthreads();
    for (int s = 128; s > 0; s >>= 1) { if (t < s) red[t] += red[t + s]; __syncthreads(); }
    const float mean = red[0] * (1.0f / DMODEL); __syncthreads();
    const float dv = v - mean;
    red[t] = dv * dv; __syncthreads();
    for (int s = 128; s > 0; s >>= 1) { if (t < s) red[t] += red[t + s]; __syncthreads(); }
    const float rstd = rsqrtf(red[0] * (1.0f / DMODEL) + 1e-5f);
    const float o = dv * rstd * g[t] + bta[t];
    Out [(size_t)m * DMODEL + t] = o;
    OutH[(size_t)m * DMODEL + t] = (_Float16)o;
}

// ---------------------------------------------------------------------------
// Causal depthwise conv1d (width 4) + bias + SiLU; dual f32 + f16 output
// ---------------------------------------------------------------------------
__global__ __launch_bounds__(256)
void conv_silu_kernel(const float* __restrict__ XZ, const float* __restrict__ cw,
                      const float* __restrict__ cb, float* __restrict__ XI,
                      _Float16* __restrict__ XIH) {
    const int idx = blockIdx.x * 256 + threadIdx.x;      // over MROWS*DINNER
    if (idx >= MROWS * DINNER) return;
    const int d = idx & (DINNER - 1);
    const int m = idx >> 9;
    const int l = m & (LL - 1);
    const int b = m >> 10;
    float s = cb[d];
#pragma unroll
    for (int j = 0; j < 4; ++j) {
        const int li = l - 3 + j;
        if (li >= 0)
            s += cw[d * 4 + j] * XZ[((size_t)(b * LL + li)) * (2 * DINNER) + d];
    }
    const float o = s / (1.0f + __expf(-s));             // silu
    XI [idx] = o;
    XIH[idx] = (_Float16)o;
}

// ---------------------------------------------------------------------------
// dt = softplus(dt_raw @ dtw^T + dtb); dt_raw = DBL[:, 0:16] (stride DBLP)
// ---------------------------------------------------------------------------
__global__ __launch_bounds__(256)
void dt_kernel(const float* __restrict__ DBL, const float* __restrict__ dtw,
               const float* __restrict__ dtb, float* __restrict__ DTo) {
    const int idx = blockIdx.x * 256 + threadIdx.x;      // over MROWS*DINNER
    if (idx >= MROWS * DINNER) return;
    const int d = idx & (DINNER - 1);
    const int m = idx >> 9;
    const float* r = DBL + (size_t)m * DBLP;
    const float* w = dtw + d * DTRANK;
    float s = dtb[d];
#pragma unroll
    for (int j = 0; j < DTRANK; ++j) s += r[j] * w[j];
    DTo[idx] = (s > 20.0f) ? s : __logf(1.0f + __expf(s));   // softplus
}

// ---------------------------------------------------------------------------
// Selective scan: one wave32 per (b,d) channel; lane owns 4 of 128 states.
// h_n = exp(dt*A_n)*h_n + dt*B_n*xi ; y = sum_n h_n*C_n (wave shfl-xor reduce)
// Fused: y = (y + xi*D) * silu(z). Output f16 (consumed only by out_proj WMMA).
// ---------------------------------------------------------------------------
__global__ __launch_bounds__(256)
void scan_kernel(const float* __restrict__ DT, const float* __restrict__ DBL,
                 const float* __restrict__ XI, const float* __restrict__ XZ,
                 const float* __restrict__ A_log, const float* __restrict__ Dp,
                 _Float16* __restrict__ Y) {
    const int lane = threadIdx.x & 31;
    const int wave = threadIdx.x >> 5;
    const int chan = blockIdx.x * 8 + wave;              // 0..2047
    const int b = chan >> 9;
    const int d = chan & (DINNER - 1);

    float Aj[4], hj[4];
#pragma unroll
    for (int j = 0; j < 4; ++j) {
        Aj[j] = -__expf(A_log[d * NSTATE + lane + 32 * j]);
        hj[j] = 0.0f;
    }
    const float Dd = Dp[d];
    const size_t base = (size_t)b * LL;
    for (int l = 0; l < LL; ++l) {
        const size_t m = base + l;
        const float* bl = DBL + m * DBLP;
        __builtin_prefetch(bl + 8 * DBLP + lane, 0, 0);  // global_prefetch_b8, 8 steps ahead
        const float dtv = DT[m * DINNER + d];
        const float xiv = XI[m * DINNER + d];
        float acc = 0.0f;
#pragma unroll
        for (int j = 0; j < 4; ++j) {
            const int n = lane + 32 * j;
            const float Bv = bl[DTRANK + n];
            const float Cv = bl[DTRANK + NSTATE + n];
            hj[j] = __expf(dtv * Aj[j]) * hj[j] + dtv * Bv * xiv;
            acc += hj[j] * Cv;
        }
#pragma unroll
        for (int off = 16; off > 0; off >>= 1) acc += __shfl_xor(acc, off, 32);
        if (lane == 0) {
            const float zv = XZ[m * (2 * DINNER) + DINNER + d];
            float yv = acc + xiv * Dd;
            yv *= zv / (1.0f + __expf(-zv));             // * silu(z)
            Y[m * DINNER + d] = (_Float16)yv;
        }
    }
}

// ---------------------------------------------------------------------------
// Head: out[m,v] = XF[m,:] . head_w[v,:] + head_b[v]  (N=20, too skinny for WMMA)
// ---------------------------------------------------------------------------
__global__ __launch_bounds__(256)
void head_kernel(const float* __restrict__ X, const float* __restrict__ hw,
                 const float* __restrict__ hb, float* __restrict__ Out) {
    const int idx = blockIdx.x * 256 + threadIdx.x;      // over MROWS*VOCAB
    if (idx >= MROWS * VOCAB) return;
    const int v = idx % VOCAB;
    const int m = idx / VOCAB;
    const float* x = X + (size_t)m * DMODEL;
    const float* w = hw + v * DMODEL;
    float s = hb[v];
#pragma unroll 8
    for (int k = 0; k < DMODEL; ++k) s += x[k] * w[k];
    Out[idx] = s;
}

extern "C" void kernel_launch(void* const* d_in, const int* in_sizes, int n_in,
                              void* d_out, int out_size, void* d_ws, size_t ws_size,
                              hipStream_t stream) {
    const int*   tokens = (const int*)d_in[0];
    const float* embed  = (const float*)d_in[1];
    const float* ning = (const float*)d_in[2];
    const float* ninb = (const float*)d_in[3];
    const float* lng[2] = {(const float*)d_in[4], (const float*)d_in[6]};  // norm1, norm_out
    const float* lnb[2] = {(const float*)d_in[5], (const float*)d_in[7]};
    const float* hw = (const float*)d_in[8];
    const float* hb = (const float*)d_in[9];
    const float *inw[2], *cw[2], *cb[2], *xw[2], *dtw[2], *dtb[2], *Alog[2], *Dp[2], *ow[2];
    for (int l = 0; l < 2; ++l) {
        const int o = 10 + l * 9;
        inw[l]  = (const float*)d_in[o + 0];
        cw[l]   = (const float*)d_in[o + 1];
        cb[l]   = (const float*)d_in[o + 2];
        xw[l]   = (const float*)d_in[o + 3];
        dtw[l]  = (const float*)d_in[o + 4];
        dtb[l]  = (const float*)d_in[o + 5];
        Alog[l] = (const float*)d_in[o + 6];
        Dp[l]   = (const float*)d_in[o + 7];
        ow[l]   = (const float*)d_in[o + 8];
    }

    float* ws = (float*)d_ws;
    float* X0  = ws;                                  // 4096*256 f32
    float* X1  = ws + 1048576;                        // 4096*256 f32
    float* XN  = ws + 2097152;                        // 4096*256 f32
    float* XZ  = ws + 3145728;                        // 4096*1024 f32
    float* XI  = ws + 7340032;                        // 4096*512 f32
    float* DBL = ws + 9437184;                        // 4096*288 f32 (padded)
    float* DT  = ws + 10616832;                       // 4096*512 f32
    _Float16* hbase = (_Float16*)(ws + 12713984);     // 16B-aligned f16 region
    _Float16* HX  = hbase;                            // 4096*256 f16 (LN out)
    _Float16* HXI = hbase + 1048576;                  // 4096*512 f16
    _Float16* HY  = hbase + 3145728;                  // 4096*512 f16
    _Float16* HW  = hbase + 5242880;                  // 262144 f16 (weight scratch)

    embed_ln_kernel<<<MROWS, 256, 0, stream>>>(tokens, embed, ning, ninb, X0, HX);

    const float* res = X0;
    float* outx[2] = {X1, X0};
    for (int l = 0; l < 2; ++l) {
        // in_proj: (4096x256)x(256x1024) -> XZ
        cvt_f16_kernel<<<(2*DINNER*DMODEL)/256, 256, 0, stream>>>(
            inw[l], HW, 2*DINNER*DMODEL, 2*DINNER*DMODEL);
        wmma_gemm_kernel<DMODEL><<<(MROWS/64)*(2*DINNER/32)/8, 256, 0, stream>>>(
            HX, HW, nullptr, XZ, MROWS, 2*DINNER);
        conv_silu_kernel<<<(MROWS*DINNER)/256, 256, 0, stream>>>(XZ, cw[l], cb[l], XI, HXI);
        // x_proj: (4096x512)x(512x272->288 padded) -> DBL (stride 288)
        cvt_f16_kernel<<<(DBLP*DINNER)/256, 256, 0, stream>>>(
            xw[l], HW, DBLW*DINNER, DBLP*DINNER);
        wmma_gemm_kernel<DINNER><<<(MROWS/64)*(DBLP/32)/8, 256, 0, stream>>>(
            HXI, HW, nullptr, DBL, MROWS, DBLP);
        dt_kernel<<<(MROWS*DINNER)/256, 256, 0, stream>>>(DBL, dtw[l], dtb[l], DT);
        scan_kernel<<<(BB*DINNER)/8, 256, 0, stream>>>(DT, DBL, XI, XZ, Alog[l], Dp[l], HY);
        // out_proj + residual: (4096x512)x(512x256) + res -> outx
        cvt_f16_kernel<<<(DMODEL*DINNER)/256, 256, 0, stream>>>(
            ow[l], HW, DMODEL*DINNER, DMODEL*DINNER);
        wmma_gemm_kernel<DINNER><<<(MROWS/64)*(DMODEL/32)/8, 256, 0, stream>>>(
            HY, HW, res, outx[l], MROWS, DMODEL);
        ln_kernel<<<MROWS, 256, 0, stream>>>(outx[l], lng[l], lnb[l], XN, HX);
        res = outx[l];
    }

    head_kernel<<<(MROWS*VOCAB + 255)/256, 256, 0, stream>>>(XN, hw, hb, (float*)d_out);
}
